// Model_89953795048038
// MI455X (gfx1250) — compile-verified, matrix operand order checked by reading, not run on hardware
//
#include <hip/hip_runtime.h>
#include <hip/hip_bf16.h>

// ---------------------------------------------------------------------------
// Persistent fused LSTM for MI455X (gfx1250, wave32, WMMA).
//   - recurrent GEMM: v_wmma_f32_16x16x32_f16
//   - W_hh slice (64KB fp16) staged ONCE into LDS, B frags via ds_load_b128
//   - cell state c kept in VGPRs across all 1024 timesteps
//   - h double-buffered in global scratch (fp16), device-wide barrier per step
//   - input GEMM (K=10) + biases from LDS, logits (N=9) fused per step
// ---------------------------------------------------------------------------

typedef __attribute__((ext_vector_type(16))) _Float16 v16h;
typedef __attribute__((ext_vector_type(8)))  _Float16 v8h;
typedef __attribute__((ext_vector_type(8)))  float    v8f;

#define BATCH   256
#define SEQ     1024
#define HIDDEN  512
#define NGATES  4
#define NCLS    9
#define INSZ    10
#define NWG     32          // one WG per 16 hidden columns (512/16)
#define NTHR    256         // 8 wave32 waves
#define WROWS   (NGATES * 16)       // 64 weight rows per WG
#define WPITCH  (HIDDEN + 8)        // +8 halves: LDS bank spread

union Frag { v16h v; v8h h[2]; };

// ---- global-address-space accessors (emit global_load/store, not flat) ----
__device__ __forceinline__ v8h gload8(const _Float16* p) {
    return *(const __attribute__((address_space(1))) v8h*)p;
}
__device__ __forceinline__ float gloadf(const float* p) {
    return *(const __attribute__((address_space(1))) float*)p;
}
__device__ __forceinline__ void gstoreh(_Float16* p, _Float16 v) {
    *(__attribute__((address_space(1))) _Float16*)p = v;
}
__device__ __forceinline__ void gstoref(float* p, float v) {
    *(__attribute__((address_space(1))) float*)p = v;
}

__device__ __forceinline__ float sigf(float x) {
    return 1.0f / (1.0f + __expf(-x));
}

// ---------------------------------------------------------------------------
// Prep: W_hh f32[2048,512] -> f16, zero h double-buffer + barrier counter
// ---------------------------------------------------------------------------
__global__ void lstm_init_kernel(const float* __restrict__ W_hh,
                                 _Float16* __restrict__ wh16,
                                 _Float16* __restrict__ hbufs,   // 2*256*512
                                 unsigned* __restrict__ counter) {
    const size_t nW = (size_t)NGATES * HIDDEN * HIDDEN;   // 1,048,576
    const size_t nH = (size_t)2 * BATCH * HIDDEN;         //   262,144
    for (size_t i = blockIdx.x * (size_t)blockDim.x + threadIdx.x;
         i < nW; i += (size_t)gridDim.x * blockDim.x)
        gstoreh(wh16 + i, (_Float16)gloadf(W_hh + i));
    for (size_t i = blockIdx.x * (size_t)blockDim.x + threadIdx.x;
         i < nH; i += (size_t)gridDim.x * blockDim.x)
        gstoreh(hbufs + i, (_Float16)0.0f);
    if (blockIdx.x == 0 && threadIdx.x == 0) *counter = 0u;
}

// ---------------------------------------------------------------------------
// Device-wide barrier (monotonic counter, one increment per WG per phase)
// ---------------------------------------------------------------------------
__device__ __forceinline__ void grid_barrier(unsigned* counter, unsigned target) {
    __threadfence();
    __syncthreads();
    if (threadIdx.x == 0) {
        atomicAdd(counter, 1u);
        while (__hip_atomic_load(counter, __ATOMIC_RELAXED,
                                 __HIP_MEMORY_SCOPE_AGENT) < target) {
            __builtin_amdgcn_s_sleep(2);
        }
    }
    __syncthreads();
}

// ---------------------------------------------------------------------------
// Persistent LSTM kernel: grid = 32 WGs x 256 threads, resident for all steps
// ---------------------------------------------------------------------------
__global__ void __launch_bounds__(NTHR, 1)
lstm_persistent_kernel(const float* __restrict__ x,       // [B,T,10]
                       const float* __restrict__ W_ih,    // [2048,10]
                       const float* __restrict__ b_ih,    // [2048]
                       const float* __restrict__ b_hh,    // [2048]
                       const float* __restrict__ lin_W,   // [9,512]
                       const float* __restrict__ lin_b,   // [9]
                       const _Float16* __restrict__ wh16, // [2048,512] f16
                       _Float16* __restrict__ hbuf0,      // [256,512] f16
                       _Float16* __restrict__ hbuf1,      // [256,512] f16
                       float* __restrict__ out,           // [B,T,9]
                       unsigned* __restrict__ counter) {
    __shared__ __align__(16) _Float16 s_whh[WROWS][WPITCH]; // 65 KB: W_hh slice
    __shared__ float s_in[BATCH][INSZ];          // 10.0 KB : x[:,t,:]
    __shared__ float s_wih[NGATES][16][INSZ];    //  2.5 KB : W_ih slice
    __shared__ float s_bias[NGATES][16];         //          b_ih + b_hh slice
    __shared__ float s_linW[NCLS][HIDDEN];       // 18.0 KB
    __shared__ float s_linb[NCLS];

    const int tid  = threadIdx.x;
    const int lane = tid & 31;
    const int wid  = tid >> 5;         // wave 0..7 -> batch rows [32w, 32w+32)
    const int nl   = lane & 15;        // N (hidden col) / M within tile
    const int kh   = lane >> 4;        // K-half select (A/B fragment layout)
    const int ht   = blockIdx.x;       // hidden-column tile 0..31

    // ---- one-time LDS staging -------------------------------------------
    // W_hh slice: 64 rows (4 gates x 16 cols) x 512 K, fp16, v8h chunks
    for (int i = tid; i < WROWS * (HIDDEN / 8); i += NTHR) {
        const int row = i >> 6;              // 0..63
        const int c8  = (i & 63) * 8;        // K offset
        const int grow = (row >> 4) * HIDDEN + ht * 16 + (row & 15);
        *(v8h*)&s_whh[row][c8] = gload8(wh16 + (size_t)grow * HIDDEN + c8);
    }
    if (tid < NGATES * 16) {
        const int g = tid >> 4, col = tid & 15;
        const int row = g * HIDDEN + ht * 16 + col;
        #pragma unroll
        for (int j = 0; j < INSZ; ++j)
            s_wih[g][col][j] = gloadf(W_ih + row * INSZ + j);
        s_bias[g][col] = gloadf(b_ih + row) + gloadf(b_hh + row);
    }
    for (int i = tid; i < NCLS * HIDDEN; i += NTHR)
        s_linW[i / HIDDEN][i % HIDDEN] = gloadf(lin_W + i);
    if (tid < NCLS) s_linb[tid] = gloadf(lin_b + tid);

    float creg[16];                    // persistent cell state (2 Mtiles x 8)
    #pragma unroll
    for (int i = 0; i < 16; ++i) creg[i] = 0.0f;

    // B-fragment LDS base pointers (invariant across timesteps)
    const _Float16* sb[NGATES];
    #pragma unroll
    for (int g = 0; g < NGATES; ++g)
        sb[g] = &s_whh[g * 16 + nl][kh * 8];

    __syncthreads();

    _Float16* bufs[2] = { hbuf0, hbuf1 };

    for (int t = 0; t <= SEQ; ++t) {
        _Float16* hb_cur  = bufs[t & 1];         // holds h_{t-1}
        _Float16* hb_next = bufs[(t + 1) & 1];   // receives h_t

        // ---- logits for step t-1 (h_{t-1} complete after last barrier) ---
        if (t > 0) {
            const int row = blockIdx.x * 8 + wid;        // 32*8 = 256 rows
            const _Float16* hp = hb_cur + (size_t)row * HIDDEN + lane * 16;
            v8h h0 = gload8(hp);
            v8h h1 = gload8(hp + 8);
            float hf[16];
            #pragma unroll
            for (int k = 0; k < 8; ++k) { hf[k] = (float)h0[k]; hf[8 + k] = (float)h1[k]; }
            #pragma unroll
            for (int c = 0; c < NCLS; ++c) {
                float s = 0.0f;
                #pragma unroll
                for (int k = 0; k < 16; ++k)
                    s += hf[k] * s_linW[c][lane * 16 + k];
                #pragma unroll
                for (int off = 16; off >= 1; off >>= 1)
                    s += __shfl_xor(s, off, 32);
                if (lane == 0)
                    gstoref(out + ((size_t)row * SEQ + (t - 1)) * NCLS + c,
                            s + s_linb[c]);
            }
        }
        if (t == SEQ) break;

        // ---- stage x[:, t, :] into LDS ----------------------------------
        for (int i = tid; i < BATCH * INSZ; i += NTHR) {
            const int b = i / INSZ, j = i - b * INSZ;
            s_in[b][j] = gloadf(x + ((size_t)b * SEQ + t) * INSZ + j);
        }
        __syncthreads();

        // ---- recurrent GEMM: h_{t-1}(256x512) @ W_hh^T slice -> 4 gates --
        // A (h) from L2 via global_load_b128; B (weights) from LDS via
        // ds_load_b128; each B frag consumed immediately, reused across both
        // M-tiles -> no spills.
        const _Float16* ab0 = hb_cur + (size_t)(wid * 32 + nl) * HIDDEN + kh * 8;
        const _Float16* ab1 = ab0 + (size_t)16 * HIDDEN;

        v8f acc[2][NGATES] = {};
        #pragma unroll 2
        for (int k0 = 0; k0 < HIDDEN; k0 += 32) {
            Frag A[2];
            A[0].h[0] = gload8(ab0 + k0);
            A[0].h[1] = gload8(ab0 + k0 + 16);
            A[1].h[0] = gload8(ab1 + k0);
            A[1].h[1] = gload8(ab1 + k0 + 16);
            if (k0 + 32 < HIDDEN) {
                __builtin_prefetch(ab0 + k0 + 32, 0, 3);
                __builtin_prefetch(ab1 + k0 + 32, 0, 3);
            }
            #pragma unroll
            for (int g = 0; g < NGATES; ++g) {
                Frag B;
                B.h[0] = *(const v8h*)(sb[g] + k0);
                B.h[1] = *(const v8h*)(sb[g] + k0 + 16);
                acc[0][g] = __builtin_amdgcn_wmma_f32_16x16x32_f16(
                    false, A[0].v, false, B.v, (short)0, acc[0][g], false, false);
                acc[1][g] = __builtin_amdgcn_wmma_f32_16x16x32_f16(
                    false, A[1].v, false, B.v, (short)0, acc[1][g], false, false);
            }
        }

        // ---- per-lane cell update: i/f/g/o live in matching acc slots ----
        #pragma unroll
        for (int m = 0; m < 2; ++m) {
            #pragma unroll
            for (int r = 0; r < 8; ++r) {
                const int b = wid * 32 + m * 16 + kh * 8 + r;
                float xg[NGATES];
                #pragma unroll
                for (int g = 0; g < NGATES; ++g) {
                    float s = s_bias[g][nl];
                    #pragma unroll
                    for (int j = 0; j < INSZ; ++j)
                        s += s_in[b][j] * s_wih[g][nl][j];
                    xg[g] = s;
                }
                const float ig = sigf(acc[m][0][r] + xg[0]);
                const float fg = sigf(acc[m][1][r] + xg[1]);
                const float gg = tanhf(acc[m][2][r] + xg[2]);
                const float og = sigf(acc[m][3][r] + xg[3]);
                const float cn = fg * creg[m * 8 + r] + ig * gg;
                creg[m * 8 + r] = cn;
                const float hv = og * tanhf(cn);
                gstoreh(hb_next + (size_t)b * HIDDEN + ht * 16 + nl,
                        (_Float16)hv);
            }
        }

        // ---- device-wide step barrier -----------------------------------
        grid_barrier(counter, (unsigned)(t + 1) * NWG);
    }
}

// ---------------------------------------------------------------------------
extern "C" void kernel_launch(void* const* d_in, const int* in_sizes, int n_in,
                              void* d_out, int out_size, void* d_ws, size_t ws_size,
                              hipStream_t stream) {
    const float* x     = (const float*)d_in[0];
    const float* W_ih  = (const float*)d_in[1];
    const float* W_hh  = (const float*)d_in[2];
    const float* b_ih  = (const float*)d_in[3];
    const float* b_hh  = (const float*)d_in[4];
    const float* lin_W = (const float*)d_in[5];
    const float* lin_b = (const float*)d_in[6];
    float* out = (float*)d_out;

    char* ws = (char*)d_ws;
    const size_t whBytes = (size_t)NGATES * HIDDEN * HIDDEN * sizeof(_Float16); // 2 MB
    const size_t hBytes  = (size_t)BATCH * HIDDEN * sizeof(_Float16);           // 256 KB
    _Float16* wh16  = (_Float16*)ws;
    _Float16* hbuf0 = (_Float16*)(ws + whBytes);
    _Float16* hbuf1 = (_Float16*)(ws + whBytes + hBytes);
    unsigned* counter = (unsigned*)(ws + whBytes + 2 * hBytes);

    lstm_init_kernel<<<1024, 256, 0, stream>>>(W_hh, wh16, hbuf0, counter);
    lstm_persistent_kernel<<<NWG, NTHR, 0, stream>>>(
        x, W_ih, b_ih, b_hh, lin_W, lin_b, wh16, hbuf0, hbuf1, out, counter);
}
